// NeighborhoodAttention2D_19439021982088
// MI455X (gfx1250) — matrix operand order repaired and checked by reading, not compiled, not run on hardware
//
#include <hip/hip_runtime.h>

typedef __attribute__((ext_vector_type(16))) __bf16 v16bf;
typedef __attribute__((ext_vector_type(8)))  float  v8f;
typedef __attribute__((ext_vector_type(4)))  unsigned int v4u;
typedef __attribute__((ext_vector_type(8)))  unsigned int v8u;

#define NPIX   25088      // 8*56*56
#define CDIM   512
#define NHEADS 16
#define HDIM   32
#define HH     56
#define WW     56
#define SCALE_Q 0.17677669529663689f   // 32^-0.5

union Frag { unsigned int u[8]; v16bf v; };

__device__ __forceinline__ unsigned short f2bf(float f) {
    union { float f; unsigned int u; } c; c.f = f;
    unsigned int u = c.u;
    unsigned int r = (u + 0x7FFFu + ((u >> 16) & 1u)) >> 16;
    return (unsigned short)r;
}

// ---------------- TDM: issue one 2-D tile load (bf16 elements) ----------------
// Builds D# group0 (4 SGPRs) + group1 (8 SGPRs) per CDNA5 ISA
// and issues TENSOR_LOAD_TO_LDS (2-group form: VADDR2/3 = NULL, tensors <= 2D).
__device__ __forceinline__ void tdm_load_2d(unsigned lds_off,
                                            unsigned long long gaddr,
                                            unsigned tensor_d0, unsigned tensor_d1,
                                            unsigned tile_d0, unsigned tile_d1,
                                            unsigned long long stride0) {
    v4u g0;
    g0[0] = 1u;                                          // count=1, user desc
    g0[1] = lds_off;                                     // lds_addr (bytes)
    g0[2] = (unsigned)(gaddr & 0xFFFFFFFFull);           // global_addr[31:0]
    g0[3] = (unsigned)((gaddr >> 32) & 0x01FFFFFFull)    // global_addr[56:32]
          | (2u << 30);                                  // type = 2 ("image")
    v8u g1;
    g1[0] = (1u << 16);                                  // data_size=1 -> 2 bytes
    g1[1] = (tensor_d0 & 0xFFFFu) << 16;                 // tensor_dim0[15:0]
    g1[2] = (tensor_d0 >> 16) | ((tensor_d1 & 0xFFFFu) << 16);
    g1[3] = (tensor_d1 >> 16) | ((tile_d0 & 0xFFFFu) << 16);
    g1[4] = (tile_d1 & 0xFFFFu);                         // tile_dim1 (tile_dim2=0)
    g1[5] = (unsigned)(stride0 & 0xFFFFFFFFull);         // tensor_dim0_stride lo
    g1[6] = (unsigned)((stride0 >> 32) & 0xFFFFull);     // stride hi (dim1 stride=0)
    g1[7] = 0u;
    asm volatile("tensor_load_to_lds %0, %1" :: "s"(g0), "s"(g1) : "memory");
}

// ---------------- conversion kernels ----------------
__global__ void cvt_f32_bf16(const float* __restrict__ in,
                             unsigned short* __restrict__ out, int n) {
    int i = blockIdx.x * blockDim.x + threadIdx.x;
    if (i < n) out[i] = f2bf(in[i]);
}

// in[K][N] (row-major) -> out[N][K] bf16 (so WMMA B fragments read contiguous K pairs)
__global__ void cvt_transpose_bf16(const float* __restrict__ in,
                                   unsigned short* __restrict__ out,
                                   int K, int N) {
    int i = blockIdx.x * blockDim.x + threadIdx.x;
    if (i < K * N) {
        int n = i / K, k = i - n * K;
        out[i] = f2bf(in[(long)k * N + n]);
    }
}

// ---------------- bf16 GEMM (K fixed = 512), TDM-staged + double-buffered ----
// A: [M x 512] bf16 row-major. Bt: [N x 512] bf16 (B transposed).
// mode 0: Cout[m*Ndim+n] = acc + bias[n]        (f32)
// mode 1: scatter into q/k/v bf16 buffers [pix][512]; q pre-scaled by SCALE_Q
__global__ __launch_bounds__(256) void gemm_bf16_kernel(
    const unsigned short* __restrict__ A,
    const unsigned short* __restrict__ Bt,
    const float* __restrict__ bias,
    float* __restrict__ Cout,
    unsigned short* __restrict__ qo,
    unsigned short* __restrict__ ko,
    unsigned short* __restrict__ vo,
    int Mdim, int Ndim, int mode)
{
    __shared__ unsigned short ash[2][32 * 32];    //  4 KB : A tiles (ping/pong)
    __shared__ unsigned short bsh[2][256 * 32];   // 32 KB : B tiles (ping/pong)

    const int lane = threadIdx.x & 31;
    const int wid  = threadIdx.x >> 5;       // 8 waves
    const int grp  = lane >> 4;              // wave32 half
    const int ln   = lane & 15;
    const int wm = wid >> 2, wn = wid & 3;   // 2x4 wave grid
    const long m0 = (long)blockIdx.x * 32;
    const int  n0 = blockIdx.y * 256;

    const unsigned aoff[2] = { (unsigned)(size_t)&ash[0][0], (unsigned)(size_t)&ash[1][0] };
    const unsigned boff[2] = { (unsigned)(size_t)&bsh[0][0], (unsigned)(size_t)&bsh[1][0] };
    const unsigned long long abase = (unsigned long long)(size_t)A;
    const unsigned long long bbase = (unsigned long long)(size_t)Bt;

    // prologue: DMA tile 0 into buffer 0
    if (wid == 0) {
        tdm_load_2d(aoff[0], abase + (unsigned long long)m0 * CDIM * 2,
                    CDIM, (unsigned)Mdim, 32, 32, CDIM);
        tdm_load_2d(boff[0], bbase + (unsigned long long)n0 * CDIM * 2,
                    CDIM, (unsigned)Ndim, 32, 256, CDIM);
        __builtin_amdgcn_s_wait_tensorcnt(0);
    }
    __syncthreads();

    v8f acc[4];
    #pragma unroll
    for (int j = 0; j < 4; ++j)
        #pragma unroll
        for (int i = 0; i < 8; ++i) acc[j][i] = 0.0f;

    const int NSTEP = CDIM / 32;   // 16
    for (int it = 0; it < NSTEP; ++it) {
        const int p = it & 1;
        // async-prefetch next K-tile into the other buffer while we compute
        if (wid == 0 && it + 1 < NSTEP) {
            const int kb = (it + 1) * 32;
            tdm_load_2d(aoff[p ^ 1], abase + ((unsigned long long)m0 * CDIM + kb) * 2,
                        CDIM, (unsigned)Mdim, 32, 32, CDIM);
            tdm_load_2d(boff[p ^ 1], bbase + ((unsigned long long)n0 * CDIM + kb) * 2,
                        CDIM, (unsigned)Ndim, 32, 256, CDIM);
        }

        // A fragment from LDS: 16x32 bf16, contiguous k = 16*grp + 2v (lane=M, VGPR holds K pairs)
        const unsigned short* ap = &ash[p][(wm * 16 + ln) * 32];
        Frag fa;
        #pragma unroll
        for (int v = 0; v < 8; ++v) {
            int kd = 2 * v + grp * 16;
            fa.u[v] = *(const unsigned int*)(ap + kd);
        }
        #pragma unroll
        for (int j = 0; j < 4; ++j) {
            const unsigned short* bp = &bsh[p][(wn * 64 + j * 16 + ln) * 32];
            Frag fb;
            #pragma unroll
            for (int v = 0; v < 8; ++v)
                fb.u[v] = *(const unsigned int*)(bp + 2 * v + grp * 16);
            acc[j] = __builtin_amdgcn_wmma_f32_16x16x32_bf16(
                false, fa.v, false, fb.v, (short)0, acc[j], false, false);
        }

        if (wid == 0 && it + 1 < NSTEP) __builtin_amdgcn_s_wait_tensorcnt(0);
        __syncthreads();
    }

    // epilogue: C layout -> lane holds N=ln, VGPR r holds M = r + 8*grp
    #pragma unroll
    for (int j = 0; j < 4; ++j) {
        #pragma unroll
        for (int r = 0; r < 8; ++r) {
            long m = m0 + wm * 16 + r + grp * 8;
            int  n = n0 + wn * 64 + j * 16 + ln;
            float val = acc[j][r] + bias[n];
            if (mode == 0) {
                Cout[m * Ndim + n] = val;
            } else {
                int which = n >> 9;          // 0=q 1=k 2=v
                int rem   = n & 511;         // head*32 + dim
                unsigned short bv = f2bf(which == 0 ? val * SCALE_Q : val);
                unsigned short* dst = (which == 0) ? qo : ((which == 1) ? ko : vo);
                dst[m * CDIM + rem] = bv;
            }
        }
    }
}

// ---------------- NATTEN attention, WMMA ----------------
// Block: 512 threads = 16 waves = 16 heads; tile = 2x8 queries (16 = WMMA M).
// Key-union for a 2x8 tile: 8 rows x 14 cols = 112 keys, padded to 128.
__device__ __forceinline__ long keypix(int b, int kk, int sh, int sw) {
    kk = (kk < 112) ? kk : 0;
    int kr = kk / 14, kc = kk - kr * 14;
    return ((long)(b * HH + sh + kr) * WW + (sw + kc));
}

__global__ __launch_bounds__(512) void natten_kernel(
    const unsigned short* __restrict__ qb,
    const unsigned short* __restrict__ kbuf,
    const unsigned short* __restrict__ vbuf,
    const float* __restrict__ rpb,
    unsigned short* __restrict__ ao)
{
    __shared__ unsigned short probs[NHEADS][16][128];   // 64 KB

    const int lane = threadIdx.x & 31;
    const int head = threadIdx.x >> 5;
    const int grp  = lane >> 4;
    const int ln   = lane & 15;
    const int b  = blockIdx.z;
    const int h0 = blockIdx.y * 2;
    const int w0 = blockIdx.x * 8;
    const int sh = min(max(h0 - 3, 0), HH - 7);
    const int sw = min(max(w0 - 3, 0), WW - 7);

    // ---- q A-fragment (lane = query M, VGPR = K pairs over head_dim) ----
    const int qm = ln;
    const int qh = h0 + (qm >> 3), qw = w0 + (qm & 7);
    const long qpix = (long)(b * HH + qh) * WW + qw;
    const unsigned short* qptr = qb + qpix * CDIM + head * HDIM;
    Frag fa;
    #pragma unroll
    for (int v = 0; v < 8; ++v) {
        int kd = 2 * v + grp * 16;
        fa.u[v] = *(const unsigned int*)(qptr + kd);
    }

    // ---- logits: 8 N-tiles of 16 keys, single K=32 WMMA each ----
    v8f lg[8];
    #pragma unroll
    for (int nt = 0; nt < 8; ++nt) {
        int kk = nt * 16 + ln;
        long kpix = keypix(b, kk, sh, sw);
        const unsigned short* kptr = kbuf + kpix * CDIM + head * HDIM;
        Frag fb;
        #pragma unroll
        for (int v = 0; v < 8; ++v)
            fb.u[v] = *(const unsigned int*)(kptr + 2 * v + grp * 16);
        v8f z;
        #pragma unroll
        for (int i = 0; i < 8; ++i) z[i] = 0.0f;
        lg[nt] = __builtin_amdgcn_wmma_f32_16x16x32_bf16(
            false, fa.v, false, fb.v, (short)0, z, false, false);
    }

    // ---- add RPB, mask invalid taps (branchless; clamped indices) ----
    #pragma unroll
    for (int nt = 0; nt < 8; ++nt) {
        int kk = nt * 16 + ln;
        bool kin = kk < 112;
        int kkc = kin ? kk : 0;
        int kr = kkc / 14, kc = kkc - kr * 14;
        int kgr = sh + kr, kgc = sw + kc;
        #pragma unroll
        for (int r = 0; r < 8; ++r) {
            int m   = r + grp * 8;
            int qh2 = h0 + (m >> 3), qw2 = w0 + (m & 7);
            int shq = min(max(qh2 - 3, 0), HH - 7);
            int swq = min(max(qw2 - 3, 0), WW - 7);
            bool valid = kin && kgr >= shq && kgr < shq + 7
                             && kgc >= swq && kgc < swq + 7;
            int bh = min(max(kgr + 6 - qh2, 0), 12);
            int bw = min(max(kgc + 6 - qw2, 0), 12);
            float rb = rpb[head * 169 + bh * 13 + bw];
            lg[nt][r] = valid ? (lg[nt][r] + rb) : -1e30f;
        }
    }

    // ---- softmax per query row (16-lane half-wave reduction + 8 N-tiles) ----
    float rinv[8];
    #pragma unroll
    for (int r = 0; r < 8; ++r) {
        float mx = -1e30f;
        #pragma unroll
        for (int nt = 0; nt < 8; ++nt) mx = fmaxf(mx, lg[nt][r]);
        #pragma unroll
        for (int off = 1; off < 16; off <<= 1)
            mx = fmaxf(mx, __shfl_xor(mx, off, 32));
        float s = 0.0f;
        #pragma unroll
        for (int nt = 0; nt < 8; ++nt) {
            float e = __expf(lg[nt][r] - mx);
            lg[nt][r] = e; s += e;
        }
        #pragma unroll
        for (int off = 1; off < 16; off <<= 1)
            s += __shfl_xor(s, off, 32);
        rinv[r] = 1.0f / s;
    }

    // ---- stage probabilities in LDS (re-layout C -> A matrix) ----
    #pragma unroll
    for (int nt = 0; nt < 8; ++nt) {
        int kk = nt * 16 + ln;
        #pragma unroll
        for (int r = 0; r < 8; ++r) {
            int m = r + grp * 8;
            probs[head][m][kk] = f2bf(lg[nt][r] * rinv[r]);
        }
    }
    __syncthreads();

    // ---- AV: attn (16x128) x V (128x32) = 4 K-chunks x 2 N-tiles ----
    v8f oacc[2];
    #pragma unroll
    for (int j = 0; j < 2; ++j)
        #pragma unroll
        for (int i = 0; i < 8; ++i) oacc[j][i] = 0.0f;

    #pragma unroll
    for (int kc4 = 0; kc4 < 4; ++kc4) {
        Frag pa;
        #pragma unroll
        for (int v = 0; v < 8; ++v) {
            int kloc = kc4 * 32 + 2 * v + grp * 16;
            pa.u[v] = *(const unsigned int*)&probs[head][ln][kloc];
        }
        #pragma unroll
        for (int nt2 = 0; nt2 < 2; ++nt2) {
            int n = nt2 * 16 + ln;
            Frag fv;
            #pragma unroll
            for (int v = 0; v < 8; ++v) {
                int kk0 = kc4 * 32 + 2 * v + grp * 16;
                long p0 = keypix(b, kk0, sh, sw);
                long p1 = keypix(b, kk0 + 1, sh, sw);
                unsigned int x0 = vbuf[p0 * CDIM + head * HDIM + n];
                unsigned int x1 = vbuf[p1 * CDIM + head * HDIM + n];
                fv.u[v] = x0 | (x1 << 16);
            }
            oacc[nt2] = __builtin_amdgcn_wmma_f32_16x16x32_bf16(
                false, pa.v, false, fv.v, (short)0, oacc[nt2], false, false);
        }
    }

    // ---- store attention output as bf16 [pix][head*32+dim] ----
    #pragma unroll
    for (int nt2 = 0; nt2 < 2; ++nt2) {
        #pragma unroll
        for (int r = 0; r < 8; ++r) {
            int m  = r + grp * 8;
            int oh = h0 + (m >> 3), ow = w0 + (m & 7);
            long opix = (long)(b * HH + oh) * WW + ow;
            ao[opix * CDIM + head * HDIM + nt2 * 16 + ln] = f2bf(oacc[nt2][r]);
        }
    }
}

// ---------------- host launcher ----------------
extern "C" void kernel_launch(void* const* d_in, const int* in_sizes, int n_in,
                              void* d_out, int out_size, void* d_ws, size_t ws_size,
                              hipStream_t stream) {
    const float* x      = (const float*)d_in[0];
    const float* w_qkv  = (const float*)d_in[1];
    const float* b_qkv  = (const float*)d_in[2];
    const float* rpb    = (const float*)d_in[3];
    const float* w_proj = (const float*)d_in[4];
    const float* b_proj = (const float*)d_in[5];
    float* out = (float*)d_out;

    char* ws = (char*)d_ws;
    const size_t S = (size_t)NPIX * CDIM * 2;          // 25088*512 bf16
    unsigned short* xb      = (unsigned short*)(ws);
    unsigned short* wqkvT   = (unsigned short*)(ws + S);
    unsigned short* wprojT  = (unsigned short*)(ws + S + 1572864);
    unsigned short* qbuf    = (unsigned short*)(ws + S + 1572864 + 524288);
    unsigned short* kbuf    = (unsigned short*)((char*)qbuf + S);
    unsigned short* vbuf    = (unsigned short*)((char*)kbuf + S);
    unsigned short* abuf    = (unsigned short*)((char*)vbuf + S);
    (void)in_sizes; (void)n_in; (void)out_size; (void)ws_size;

    // 1) conversions
    {
        int n = NPIX * CDIM;
        cvt_f32_bf16<<<(n + 255) / 256, 256, 0, stream>>>(x, xb, n);
        int nq = CDIM * 3 * CDIM;
        cvt_transpose_bf16<<<(nq + 255) / 256, 256, 0, stream>>>(w_qkv, wqkvT, CDIM, 3 * CDIM);
        int np = CDIM * CDIM;
        cvt_transpose_bf16<<<(np + 255) / 256, 256, 0, stream>>>(w_proj, wprojT, CDIM, CDIM);
    }

    // 2) QKV GEMM: 25088x1536 = xb @ w_qkv + b, scatter q(scaled)/k/v bf16
    {
        dim3 grid(NPIX / 32, (3 * CDIM) / 256);
        gemm_bf16_kernel<<<grid, 256, 0, stream>>>(
            xb, wqkvT, b_qkv, nullptr, qbuf, kbuf, vbuf, NPIX, 3 * CDIM, 1);
    }

    // 3) neighborhood attention (all 16 heads per block)
    {
        dim3 grid(WW / 8, HH / 2, 8);
        natten_kernel<<<grid, 512, 0, stream>>>(qbuf, kbuf, vbuf, rpb, abuf);
    }

    // 4) output projection: 25088x512 = abuf @ w_proj + b_proj -> f32 out
    {
        dim3 grid(NPIX / 32, CDIM / 256);
        gemm_bf16_kernel<<<grid, 256, 0, stream>>>(
            abuf, wprojT, b_proj, out, nullptr, nullptr, nullptr, NPIX, CDIM, 0);
    }
}
